// Model_90168543412445
// MI455X (gfx1250) — matrix operand order repaired and checked by reading, not compiled, not run on hardware
//
#include <hip/hip_runtime.h>
#include <math.h>
#include <stdint.h>

// ---------------- types (CDNA5 / gfx1250, wave32) ----------------
typedef _Float16 h16;
typedef _Float16 hvec16 __attribute__((ext_vector_type(16)));
typedef _Float16 hvec8  __attribute__((ext_vector_type(8)));
typedef float    fvec8  __attribute__((ext_vector_type(8)));

#define TOPK 20      // int(3*log(1024)) = 20
#define BM 256       // block tile rows  (4 row-waves x 64)
#define BN 128       // block tile cols  (2 col-waves x 64)
#define BK 32
#define LDT 40       // LDS row stride in halfs (32 + 8 pad -> 80B rows, bank-rotating)

__device__ inline int clampi(int v, int lo, int hi) { return v < lo ? lo : (v > hi ? hi : v); }

// ================= GEMM-NT: C = act(A[MxK] * Bn[NxK]^T + bias) (+resid) ==========
// A, Bn, C, resid are f16 in memory; accumulation f32 via WMMA (16 per K-step/wave).
// Tiles staged global->LDS with async-to-LDS (ASYNCcnt) + double buffering;
// synchronous guarded fallback for ragged tiles (K%8!=0 or partial M/N blocks).
__global__ void __launch_bounds__(256)
gemm_nt_kernel(const h16* __restrict__ A, const h16* __restrict__ Bn,
               const float* __restrict__ bias, const h16* __restrict__ resid,
               h16* __restrict__ C, int M, int N, int K,
               long long sA, long long sB, long long sC, int act)
{
    __shared__ h16 As[2][BM * LDT];
    __shared__ h16 Bs[2][BN * LDT];

    const int tid  = threadIdx.x;
    const int lane = tid & 31;
    const int wave = tid >> 5;
    const int wm   = wave >> 1;  // 0..3 -> 64-row slice
    const int wn   = wave & 1;   // 0..1 -> 64-col slice
    const int bRow = blockIdx.y * BM;
    const int bCol = blockIdx.x * BN;
    const long long z = blockIdx.z;

    const h16* Ab = A + z * sA;
    const h16* Bb = Bn + z * sB;
    h16*       Cb = C + z * sC;
    const h16* Rb = resid ? (resid + z * sC) : nullptr;

    fvec8 zero8 = {};
    fvec8 acc[4][4];
    for (int i = 0; i < 4; ++i)
        for (int j = 0; j < 4; ++j) acc[i][j] = zero8;

    const int  KT    = (K + BK - 1) / BK;
    const bool k8    = ((K & 7) == 0);
    const bool aFull = (bRow + BM <= M);
    const bool bFull = (bCol + BN <= N);

    // warm GL2 for the first tiles (global_prefetch_b8; speculative, safe)
    __builtin_prefetch(Ab + (long long)bRow * K, 0, 1);
    __builtin_prefetch(Bb + (long long)bCol * K, 0, 1);

    // stage one BK-slice of A and B into LDS buffer `pbuf`; returns per-wave async count
    auto stage = [&](int kt, int pbuf) -> int {
        const int k0 = kt * BK;
        const bool kOK = k8 && (k0 + BK <= K);
        int cnt = 0;
        // ---- A tile (BM x BK), 1024 16B-chunks ----
        if (kOK && aFull) {
            for (int qq = tid; qq < 1024; qq += 256) {
                int row = qq >> 2, c8 = (qq & 3) * 8;
                unsigned long long g =
                    (unsigned long long)(uintptr_t)(Ab + (long long)(bRow + row) * K + k0 + c8);
                unsigned int l =
                    (unsigned int)(unsigned long long)(uintptr_t)(&As[pbuf][row * LDT + c8]);
                asm volatile("global_load_async_to_lds_b128 %0, %1, off"
                             :: "v"(l), "v"(g) : "memory");
            }
            cnt += 4;
        } else {
            for (int qq = tid; qq < 1024; qq += 256) {
                int row = qq >> 2, c8 = (qq & 3) * 8;
                int gr = bRow + row;
                hvec8 hv;
                for (int e = 0; e < 8; ++e) {
                    int gk = k0 + c8 + e;
                    hv[e] = (gr < M && gk < K) ? Ab[(long long)gr * K + gk] : (h16)0.f;
                }
                *(hvec8*)(&As[pbuf][row * LDT + c8]) = hv;
            }
        }
        // ---- B tile (BN x BK), 512 16B-chunks ----
        if (kOK && bFull) {
            for (int qq = tid; qq < 512; qq += 256) {
                int row = qq >> 2, c8 = (qq & 3) * 8;
                unsigned long long g =
                    (unsigned long long)(uintptr_t)(Bb + (long long)(bCol + row) * K + k0 + c8);
                unsigned int l =
                    (unsigned int)(unsigned long long)(uintptr_t)(&Bs[pbuf][row * LDT + c8]);
                asm volatile("global_load_async_to_lds_b128 %0, %1, off"
                             :: "v"(l), "v"(g) : "memory");
            }
            cnt += 2;
        } else {
            for (int qq = tid; qq < 512; qq += 256) {
                int row = qq >> 2, c8 = (qq & 3) * 8;
                int gn = bCol + row;
                hvec8 hv;
                for (int e = 0; e < 8; ++e) {
                    int gk = k0 + c8 + e;
                    hv[e] = (gn < N && gk < K) ? Bb[(long long)gn * K + gk] : (h16)0.f;
                }
                *(hvec8*)(&Bs[pbuf][row * LDT + c8]) = hv;
            }
        }
        return cnt;
    };
    auto waitAsync = [&](int n) {  // ASYNCcnt completes in order
        if (n == 0)      asm volatile("s_wait_asynccnt 0x0" ::: "memory");
        else if (n == 2) asm volatile("s_wait_asynccnt 0x2" ::: "memory");
        else if (n == 4) asm volatile("s_wait_asynccnt 0x4" ::: "memory");
        else             asm volatile("s_wait_asynccnt 0x6" ::: "memory");
    };

    stage(0, 0);
    for (int kt = 0; kt < KT; ++kt) {
        int inflightNext = 0;
        if (kt + 1 < KT) inflightNext = stage(kt + 1, (kt + 1) & 1);
        waitAsync(inflightNext);   // drain everything except next-stage asyncs
        __syncthreads();

        const h16* Abase = As[kt & 1];
        const h16* Bbase = Bs[kt & 1];
        const int half = lane >> 4;
        const int lr   = lane & 15;
        hvec16 afr[4], bfr[4];
        for (int i = 0; i < 4; ++i) {   // A frag: interleaved 8-half K-blocks
            const h16* p = &Abase[(wm * 64 + i * 16 + lr) * LDT];
            hvec8 lo = *(const hvec8*)(p + half * 8);
            hvec8 hi = *(const hvec8*)(p + 16 + half * 8);
            for (int e = 0; e < 8; ++e) { afr[i][e] = lo[e]; afr[i][8 + e] = hi[e]; }
        }
        for (int j = 0; j < 4; ++j) {   // B frag: interleaved 8-half K-blocks
            const h16* p = &Bbase[(wn * 64 + j * 16 + lr) * LDT];
            hvec8 lo = *(const hvec8*)(p + half * 8);
            hvec8 hi = *(const hvec8*)(p + 16 + half * 8);
            for (int e = 0; e < 8; ++e) { bfr[j][e] = lo[e]; bfr[j][8 + e] = hi[e]; }
        }
        for (int i = 0; i < 4; ++i)
            for (int j = 0; j < 4; ++j)
                acc[i][j] = __builtin_amdgcn_wmma_f32_16x16x32_f16(
                    false, afr[i], false, bfr[j], (short)0, acc[i][j], false, false);
        __syncthreads();   // all waves done reading buf[kt&1] before it is restaged
    }

    // ---- epilogue: bias, GELU, residual, bounds-guarded f16 store ----
    const int half = lane >> 4;
    const int lr   = lane & 15;
    for (int j = 0; j < 4; ++j) {
        int col = bCol + wn * 64 + j * 16 + lr;
        if (col >= N) continue;
        float bv = bias ? bias[col] : 0.f;
        for (int i = 0; i < 4; ++i) {
            for (int e = 0; e < 8; ++e) {
                int row = bRow + wm * 64 + i * 16 + half * 8 + e;
                if (row >= M) continue;
                float v = acc[i][j][e] + bv;
                if (act == 1) v = 0.5f * v * (1.0f + erff(v * 0.70710678118654752f));
                long long idx = (long long)row * N + col;
                if (Rb) v += (float)Rb[idx];
                Cb[idx] = (h16)v;
            }
        }
    }
}

// ================= helper kernels (f16 storage, f32 math) =================
__global__ void k_transpose_h(const float* __restrict__ src, h16* __restrict__ dst, int R, int C)
{
    long long i = (long long)blockIdx.x * 256 + threadIdx.x;
    if (i >= (long long)R * C) return;
    int r = (int)(i / C), c = (int)(i % C);
    dst[(long long)c * R + r] = (h16)src[i];
}

__global__ void k_cast_h(const float* __restrict__ s, h16* __restrict__ d, long long n)
{
    long long i = (long long)blockIdx.x * 256 + threadIdx.x;
    if (i < n) d[i] = (h16)s[i];
}

__global__ void k_zero(float* __restrict__ p, long long n)
{
    long long i = (long long)blockIdx.x * 256 + threadIdx.x;
    if (i < n) p[i] = 0.f;
}

__global__ void k_copy_h(const h16* __restrict__ s, h16* __restrict__ d, long long n)
{
    long long i = (long long)blockIdx.x * 256 + threadIdx.x;
    if (i < n) d[i] = s[i];
}

__global__ void k_add3_h(const h16* __restrict__ a, const h16* __restrict__ b,
                         const h16* __restrict__ c, h16* __restrict__ o, long long n)
{
    long long i = (long long)blockIdx.x * 256 + threadIdx.x;
    if (i < n) o[i] = (h16)((float)a[i] + (float)b[i] + (float)c[i]);
}

// series decomposition: moving average k=25 with edge replication (per (b,c) column).
__global__ void k_decomp(const h16* __restrict__ x, h16* __restrict__ seas,
                         h16* __restrict__ trend, int Bc, int L, int C)
{
    long long i = (long long)blockIdx.x * 256 + threadIdx.x;
    if (i >= (long long)Bc * C) return;
    int b = (int)(i / C), c = (int)(i % C);
    const long long base = (long long)b * L * C + c;
    float sum = 0.f;
    for (int j = -12; j <= 12; ++j) sum += (float)x[base + (long long)clampi(j, 0, L - 1) * C];
    for (int t = 0; t < L; ++t) {
        float m  = sum * (1.0f / 25.0f);
        float xv = (float)x[base + (long long)t * C];
        if (seas)  seas[base + (long long)t * C]  = (h16)(xv - m);
        if (trend) trend[base + (long long)t * C] = (h16)m;
        sum += (float)x[base + (long long)clampi(t + 13, 0, L - 1) * C]
             - (float)x[base + (long long)clampi(t - 12, 0, L - 1) * C];
    }
}

__global__ void k_time_mean(const h16* __restrict__ x, float* __restrict__ out,
                            int Bc, int L, int C)
{
    long long i = (long long)blockIdx.x * 256 + threadIdx.x;
    if (i >= (long long)Bc * C) return;
    int b = (int)(i / C), c = (int)(i % C);
    float s = 0.f;
    for (int t = 0; t < L; ++t) s += (float)x[((long long)b * L + t) * C + c];
    out[i] = s / (float)L;
}

// seasonal_init = [seas[:, -512:], 0]; trend_init = [trend[:, -512:], mean]
__global__ void k_dec_init(const h16* __restrict__ seasE, const h16* __restrict__ trendE,
                           const float* __restrict__ meanE, h16* __restrict__ sInit,
                           h16* __restrict__ tInit, int Bc, int L, int C)
{
    long long i = (long long)blockIdx.x * 256 + threadIdx.x;
    if (i >= (long long)Bc * L * C) return;
    int c = (int)(i % C);
    int t = (int)((i / C) % L);
    int b = (int)(i / ((long long)C * L));
    h16 sv, tv;
    if (t < 512) {
        long long src = ((long long)b * L + 512 + t) * C + c;
        sv = seasE[src]; tv = trendE[src];
    } else {
        sv = (h16)0.f; tv = (h16)meanE[b * C + c];
    }
    sInit[i] = sv; tInit[i] = tv;
}

// circular im2col, taps j=0..2 -> x[(t-1+j) mod L]
__global__ void k_im2col3(const h16* __restrict__ x, h16* __restrict__ out,
                          int Bc, int L, int C)
{
    long long i = (long long)blockIdx.x * 256 + threadIdx.x;
    long long n = (long long)Bc * L * 3 * C;
    if (i >= n) return;
    int c = (int)(i % C);
    int j = (int)((i / C) % 3);
    int t = (int)((i / ((long long)3 * C)) % L);
    int b = (int)(i / ((long long)3 * C * L));
    int ts = t - 1 + j; if (ts < 0) ts += L; if (ts >= L) ts -= L;
    out[i] = x[((long long)b * L + ts) * C + c];
}

// R[b,tau] = (1/Dm) * sum_t M[b, t, (t+tau)%L]  (M = K * Q^T), L == 1024
__global__ void __launch_bounds__(256)
k_corr_mean(const h16* __restrict__ Mc, float* __restrict__ R, int L, int Dm)
{
    int b = blockIdx.x;
    const h16* Mb = Mc + (long long)b * L * L;
    float acc0 = 0.f, acc1 = 0.f, acc2 = 0.f, acc3 = 0.f;
    for (int t = 0; t < L; ++t) {
        const h16* row = Mb + (long long)t * L;
        int c0 = t + threadIdx.x;        if (c0 >= L) c0 -= L;
        int c1 = t + threadIdx.x + 256;  if (c1 >= L) c1 -= L;
        int c2 = t + threadIdx.x + 512;  if (c2 >= L) c2 -= L;
        int c3 = t + threadIdx.x + 768;  if (c3 >= L) c3 -= L;
        acc0 += (float)row[c0]; acc1 += (float)row[c1];
        acc2 += (float)row[c2]; acc3 += (float)row[c3];
    }
    float inv = 1.0f / (float)Dm;
    R[(long long)b * L + threadIdx.x      ] = acc0 * inv;
    R[(long long)b * L + threadIdx.x + 256] = acc1 * inv;
    R[(long long)b * L + threadIdx.x + 512] = acc2 * inv;
    R[(long long)b * L + threadIdx.x + 768] = acc3 * inv;
}

// top-20 of R[b,:] (iterative argmax), then softmax over the 20 values.
__global__ void __launch_bounds__(256)
k_topk(const float* __restrict__ R, float* __restrict__ w, int* __restrict__ dly, int L)
{
    int b = blockIdx.x, tid = threadIdx.x;
    __shared__ float sv[1024];
    __shared__ float best[256];
    __shared__ int   bidx[256];
    __shared__ float topv[TOPK];
    __shared__ int   topi[TOPK];
    for (int i = tid; i < L; i += 256) sv[i] = R[(long long)b * L + i];
    __syncthreads();
    for (int it = 0; it < TOPK; ++it) {
        float bv = -3.0e38f; int bi = 0;
        for (int i = tid; i < L; i += 256) { float v = sv[i]; if (v > bv) { bv = v; bi = i; } }
        best[tid] = bv; bidx[tid] = bi;
        __syncthreads();
        for (int s = 128; s > 0; s >>= 1) {
            if (tid < s && best[tid + s] > best[tid]) { best[tid] = best[tid + s]; bidx[tid] = bidx[tid + s]; }
            __syncthreads();
        }
        if (tid == 0) { topv[it] = best[0]; topi[it] = bidx[0]; sv[bidx[0]] = -3.0e38f; }
        __syncthreads();
    }
    if (tid == 0) {
        float mx = topv[0];
        for (int i = 1; i < TOPK; ++i) mx = fmaxf(mx, topv[i]);
        float se = 0.f, ev[TOPK];
        for (int i = 0; i < TOPK; ++i) { ev[i] = expf(topv[i] - mx); se += ev[i]; }
        for (int i = 0; i < TOPK; ++i) { w[b * TOPK + i] = ev[i] / se; dly[b * TOPK + i] = topi[i]; }
    }
}

// agg[b,t,c] = sum_i w[b,i] * v[b,(t+d_i)%L,c]   (V = 16MB f16, lives in 192MB L2)
__global__ void k_aggregate(const h16* __restrict__ v, const float* __restrict__ w,
                            const int* __restrict__ dly, h16* __restrict__ out,
                            int Bc, int L, int C)
{
    long long i = (long long)blockIdx.x * 256 + threadIdx.x;
    long long n = (long long)Bc * L * C;
    if (i >= n) return;
    int c = (int)(i % C);
    int t = (int)((i / C) % L);
    int b = (int)(i / ((long long)C * L));
    float s = 0.f;
    for (int j = 0; j < TOPK; ++j) {
        int ts = t + dly[b * TOPK + j]; if (ts >= L) ts -= L;
        s += w[b * TOPK + j] * (float)v[((long long)b * L + ts) * C + c];
    }
    out[i] = (h16)s;
}

// per-(b,t)-row layernorm * g + b
__global__ void __launch_bounds__(256)
k_ln_row(const h16* __restrict__ x, const float* __restrict__ g,
         const float* __restrict__ bp, h16* __restrict__ y, int D)
{
    __shared__ float s1[256], s2[256];
    long long row = blockIdx.x;
    const h16* xr = x + row * D;
    float a = 0.f, b = 0.f;
    for (int i = threadIdx.x; i < D; i += 256) { float v = (float)xr[i]; a += v; b += v * v; }
    s1[threadIdx.x] = a; s2[threadIdx.x] = b;
    __syncthreads();
    for (int s = 128; s > 0; s >>= 1) {
        if (threadIdx.x < s) { s1[threadIdx.x] += s1[threadIdx.x + s]; s2[threadIdx.x] += s2[threadIdx.x + s]; }
        __syncthreads();
    }
    float mu  = s1[0] / (float)D;
    float var = s2[0] / (float)D - mu * mu;
    float inv = rsqrtf(var + 1e-5f);
    h16* yr = y + row * D;
    for (int i = threadIdx.x; i < D; i += 256)
        yr[i] = (h16)(((float)xr[i] - mu) * inv * g[i] + bp[i]);
}

// subtract per-(b,c) mean over the time axis (axis=1)
__global__ void k_sub_colmean(h16* __restrict__ y, int Bc, int L, int C)
{
    long long i = (long long)blockIdx.x * 256 + threadIdx.x;
    if (i >= (long long)Bc * C) return;
    int b = (int)(i / C), c = (int)(i % C);
    h16* p = y + (long long)b * L * C + c;
    float s = 0.f;
    for (int t = 0; t < L; ++t) s += (float)p[(long long)t * C];
    float m = s / (float)L;
    for (int t = 0; t < L; ++t) p[(long long)t * C] = (h16)((float)p[(long long)t * C] - m);
}

// out[b,t,c] += scale * (s[b,512+t,c] + tr[b,512+t,c]),  t in [0,512)
__global__ void k_accum_out(const h16* __restrict__ s, const h16* __restrict__ tr,
                            float* __restrict__ out, int Bc, int Lp, int C, int Lfull, float scale)
{
    long long i = (long long)blockIdx.x * 256 + threadIdx.x;
    if (i >= (long long)Bc * Lp * C) return;
    int c = (int)(i % C);
    int t = (int)((i / C) % Lp);
    int b = (int)(i / ((long long)C * Lp));
    long long src = ((long long)b * Lfull + 512 + t) * C + c;
    out[i] += scale * ((float)s[src] + (float)tr[src]);
}

// ================= host orchestration =================
extern "C" void kernel_launch(void* const* d_in, const int* in_sizes, int n_in,
                              void* d_out, int out_size, void* d_ws, size_t ws_size,
                              hipStream_t stream)
{
    (void)in_sizes; (void)n_in; (void)ws_size;
    const int Bc = 16, L = 1024, Dm = 512, DFF = 2048, CIN = 21, MARKN = 4;
    const long long NT = (long long)Bc * L;
    const long long ND = NT * Dm;

    auto F = [&](int i) -> const float* { return (const float*)d_in[i]; };
    const float* x_enc      = F(0);
    const float* x_mark_enc = F(1);
    const float* x_mark_dec = F(3);
    const float* dec_conv_w = F(4);  // (3,21,512) = (63,512)
    const float* dec_time_w = F(5);  // (4,512)

    struct AttnP { const float *bk, *bo, *bq, *bv, *wk, *wo, *wq, *wv; };
    struct DecP  { const float* c1; const float* c2; AttnP cross, self;
                   const float* tw; const float *nb, *ng, *pb, *pw; };
    struct EncL  { AttnP a; const float* c1; const float* c2; };
    struct EncP  { EncL l[2]; const float *nb, *ng; };

    DecP dec[2]; EncP enc[2];
    for (int d = 0; d < 2; ++d) {        // sorted leaves of decoders[d]
        int o = 6 + d * 23;
        dec[d].c1 = F(o + 0); dec[d].c2 = F(o + 1);
        dec[d].cross = { F(o+2), F(o+3), F(o+4), F(o+5), F(o+6), F(o+7), F(o+8), F(o+9) };
        dec[d].self  = { F(o+10),F(o+11),F(o+12),F(o+13),F(o+14),F(o+15),F(o+16),F(o+17) };
        dec[d].tw = F(o+18); dec[d].nb = F(o+19); dec[d].ng = F(o+20);
        dec[d].pb = F(o+21); dec[d].pw = F(o+22);
    }
    const float* enc_conv_w = F(52);
    const float* enc_time_w = F(53);
    for (int e = 0; e < 2; ++e) {
        int o = 54 + e * 22;
        for (int l = 0; l < 2; ++l) {
            int lo = o + l * 10;
            enc[e].l[l].a = { F(lo+0),F(lo+1),F(lo+2),F(lo+3),F(lo+4),F(lo+5),F(lo+6),F(lo+7) };
            enc[e].l[l].c1 = F(lo+8); enc[e].l[l].c2 = F(lo+9);
        }
        enc[e].nb = F(o + 20); enc[e].ng = F(o + 21);
    }

    // ---- workspace bump allocator ----
    size_t off = 0;
    auto allocF = [&](size_t n) -> float* {
        off = (off + 255) & ~(size_t)255;
        float* p = (float*)((char*)d_ws + off); off += n * sizeof(float); return p;
    };
    auto allocH = [&](size_t n) -> h16* {
        off = (off + 255) & ~(size_t)255;
        h16* p = (h16*)((char*)d_ws + off); off += n * sizeof(h16); return p;
    };
    auto allocI = [&](size_t n) -> int* {
        off = (off + 255) & ~(size_t)255;
        int* p = (int*)((char*)d_ws + off); off += n * sizeof(int); return p;
    };
    auto gridFor = [](long long n) { return dim3((unsigned)((n + 255) / 256)); };
    auto transp = [&](const float* src, int R, int C) -> h16* {   // fp32 (R,C) -> f16 (C,R)
        h16* dst = allocH((size_t)R * C);
        k_transpose_h<<<gridFor((long long)R * C), 256, 0, stream>>>(src, dst, R, C);
        return dst;
    };

    // ---- transpose+cast all weights once: Bn layout = (N, K) f16 ----
    h16* ecwT = transp(enc_conv_w, 3 * CIN, Dm);  // (512,63)
    h16* etwT = transp(enc_time_w, MARKN, Dm);    // (512,4)
    h16* dcwT = transp(dec_conv_w, 3 * CIN, Dm);
    h16* dtwT = transp(dec_time_w, MARKN, Dm);
    h16* eWT[2][2][4]; h16* eC1T[2][2]; h16* eC2T[2][2];
    for (int e = 0; e < 2; ++e)
        for (int l = 0; l < 2; ++l) {
            eWT[e][l][0] = transp(enc[e].l[l].a.wq, Dm, Dm);
            eWT[e][l][1] = transp(enc[e].l[l].a.wk, Dm, Dm);
            eWT[e][l][2] = transp(enc[e].l[l].a.wv, Dm, Dm);
            eWT[e][l][3] = transp(enc[e].l[l].a.wo, Dm, Dm);
            eC1T[e][l]   = transp(enc[e].l[l].c1, Dm, DFF);
            eC2T[e][l]   = transp(enc[e].l[l].c2, DFF, Dm);
        }
    h16* dSelfT[2][4]; h16* dCrossT[2][4];
    h16* dC1T[2]; h16* dC2T[2]; h16* dTwT[2]; h16* dPrT[2];
    for (int d = 0; d < 2; ++d) {
        dSelfT[d][0] = transp(dec[d].self.wq, Dm, Dm);
        dSelfT[d][1] = transp(dec[d].self.wk, Dm, Dm);
        dSelfT[d][2] = transp(dec[d].self.wv, Dm, Dm);
        dSelfT[d][3] = transp(dec[d].self.wo, Dm, Dm);
        dCrossT[d][0] = transp(dec[d].cross.wq, Dm, Dm);
        dCrossT[d][1] = transp(dec[d].cross.wk, Dm, Dm);
        dCrossT[d][2] = transp(dec[d].cross.wv, Dm, Dm);
        dCrossT[d][3] = transp(dec[d].cross.wv, Dm, Dm);
        dCrossT[d][3] = transp(dec[d].cross.wo, Dm, Dm);
        dC1T[d] = transp(dec[d].c1, Dm, DFF);
        dC2T[d] = transp(dec[d].c2, DFF, Dm);
        dTwT[d] = transp(dec[d].tw, 3 * Dm, CIN);   // (21,1536)
        dPrT[d] = transp(dec[d].pw, Dm, CIN);       // (21,512)
    }

    // ---- f16 copies of raw inputs that feed GEMMs / elementwise chain ----
    h16* xench = allocH((size_t)NT * CIN);
    h16* xmeh  = allocH((size_t)NT * MARKN);
    h16* xmdh  = allocH((size_t)NT * MARKN);
    k_cast_h<<<gridFor(NT * (long long)CIN), 256, 0, stream>>>(x_enc, xench, NT * (long long)CIN);
    k_cast_h<<<gridFor(NT * (long long)MARKN), 256, 0, stream>>>(x_mark_enc, xmeh, NT * (long long)MARKN);
    k_cast_h<<<gridFor(NT * (long long)MARKN), 256, 0, stream>>>(x_mark_dec, xmdh, NT * (long long)MARKN);

    // ---- activation buffers (f16) ----
    h16* q    = allocH(ND); h16* kk  = allocH(ND); h16* vv = allocH(ND);
    h16* agg  = allocH(ND);
    h16* corrM = allocH((size_t)Bc * L * L);
    float* Rbuf = allocF((size_t)Bc * L);
    float* wts  = allocF((size_t)Bc * TOPK);
    int*   dly  = allocI((size_t)Bc * TOPK);
    h16* eo    = allocH(ND); h16* dec0 = allocH(ND);
    h16* xbuf  = allocH(ND); h16* xb2  = allocH(ND);
    h16* encOut = allocH(ND); h16* lnT = allocH(ND);
    h16* t1 = allocH(ND); h16* t2 = allocH(ND); h16* t3 = allocH(ND);
    h16* big = allocH((size_t)NT * DFF);   // FFN mid & im2col staging
    h16* seasE = allocH((size_t)NT * CIN); h16* trendE = allocH((size_t)NT * CIN);
    float* meanE = allocF((size_t)Bc * CIN);
    h16* sInit = allocH((size_t)NT * CIN); h16* tInit = allocH((size_t)NT * CIN);
    h16* trA = allocH((size_t)NT * CIN);   h16* trB = allocH((size_t)NT * CIN);
    h16* sOut = allocH((size_t)NT * CIN);

    auto gemm = [&](const h16* A, const h16* Bn, const float* bias, const h16* resid,
                    h16* C, int M, int N, int K, int batch,
                    long long sA, long long sB, long long sC, int act) {
        dim3 g((N + BN - 1) / BN, (M + BM - 1) / BM, batch);
        gemm_nt_kernel<<<g, 256, 0, stream>>>(A, Bn, bias, resid, C, M, N, K, sA, sB, sC, act);
    };
    auto decompose = [&](const h16* x, h16* seas, h16* trend, int C) {
        k_decomp<<<gridFor((long long)Bc * C), 256, 0, stream>>>(x, seas, trend, Bc, L, C);
    };
    auto myln = [&](const h16* x, const float* g_, const float* b_, h16* y) {
        k_ln_row<<<(unsigned)NT, 256, 0, stream>>>(x, g_, b_, y, Dm);
        k_sub_colmean<<<gridFor((long long)Bc * Dm), 256, 0, stream>>>(y, Bc, L, Dm);
    };
    auto attn = [&](const h16* xq, const h16* xkv, h16* const* wT, const AttnP& p, h16* out) {
        gemm(xq,  wT[0], p.bq, nullptr, q,  (int)NT, Dm, Dm, 1, 0, 0, 0, 0);
        gemm(xkv, wT[1], p.bk, nullptr, kk, (int)NT, Dm, Dm, 1, 0, 0, 0, 0);
        gemm(xkv, wT[2], p.bv, nullptr, vv, (int)NT, Dm, Dm, 1, 0, 0, 0, 0);
        // M[b] = K_b * Q_b^T (batched), then wrapped-diagonal mean -> R[b,tau]
        gemm(kk, q, nullptr, nullptr, corrM, L, L, Dm, Bc,
             (long long)L * Dm, (long long)L * Dm, (long long)L * L, 0);
        k_corr_mean<<<Bc, 256, 0, stream>>>(corrM, Rbuf, L, Dm);
        k_topk<<<Bc, 256, 0, stream>>>(Rbuf, wts, dly, L);
        k_aggregate<<<gridFor(ND), 256, 0, stream>>>(vv, wts, dly, agg, Bc, L, Dm);
        gemm(agg, wT[3], p.bo, xq /*residual*/, out, (int)NT, Dm, Dm, 1, 0, 0, 0, 0);
    };

    // ---- output accumulator starts at zero ----
    k_zero<<<gridFor(out_size), 256, 0, stream>>>((float*)d_out, out_size);

    // ---- decomposition init + embeddings ----
    decompose(xench, seasE, trendE, CIN);
    k_time_mean<<<gridFor((long long)Bc * CIN), 256, 0, stream>>>(xench, meanE, Bc, L, CIN);
    k_dec_init<<<gridFor(NT * (long long)CIN), 256, 0, stream>>>(
        seasE, trendE, meanE, sInit, tInit, Bc, L, CIN);

    k_im2col3<<<gridFor(NT * (long long)(3 * CIN)), 256, 0, stream>>>(xench, big, Bc, L, CIN);
    gemm(big, ecwT, nullptr, nullptr, xbuf, (int)NT, Dm, 3 * CIN, 1, 0, 0, 0, 0);
    gemm(xmeh, etwT, nullptr, xbuf, eo, (int)NT, Dm, MARKN, 1, 0, 0, 0, 0);

    k_im2col3<<<gridFor(NT * (long long)(3 * CIN)), 256, 0, stream>>>(sInit, big, Bc, L, CIN);
    gemm(big, dcwT, nullptr, nullptr, xbuf, (int)NT, Dm, 3 * CIN, 1, 0, 0, 0, 0);
    gemm(xmdh, dtwT, nullptr, xbuf, dec0, (int)NT, Dm, MARKN, 1, 0, 0, 0, 0);

    for (int li = 0; li < 2; ++li) {    // N_LEARNER
        // -------- encoder --------
        k_copy_h<<<gridFor(ND), 256, 0, stream>>>(eo, xbuf, ND);
        for (int l = 0; l < 2; ++l) {
            attn(xbuf, xbuf, eWT[li][l], enc[li].l[l].a, xb2);
            decompose(xb2, xbuf, nullptr, Dm);
            gemm(xbuf, eC1T[li][l], nullptr, nullptr, big, (int)NT, DFF, Dm, 1, 0, 0, 0, 1);
            gemm(big, eC2T[li][l], nullptr, xbuf, xb2, (int)NT, Dm, DFF, 1, 0, 0, 0, 0);
            decompose(xb2, xbuf, nullptr, Dm);
        }
        myln(xbuf, enc[li].ng, enc[li].nb, encOut);

        // -------- decoder (1 layer) --------
        k_copy_h<<<gridFor(ND), 256, 0, stream>>>(dec0, xbuf, ND);
        k_copy_h<<<gridFor(NT * (long long)CIN), 256, 0, stream>>>(tInit, trA, NT * (long long)CIN);
        {
            attn(xbuf, xbuf, dSelfT[li], dec[li].self, xb2);
            decompose(xb2, xbuf, t1, Dm);
            attn(xbuf, encOut, dCrossT[li], dec[li].cross, xb2);
            decompose(xb2, xbuf, t2, Dm);
            gemm(xbuf, dC1T[li], nullptr, nullptr, big, (int)NT, DFF, Dm, 1, 0, 0, 0, 1);
            gemm(big, dC2T[li], nullptr, xbuf, xb2, (int)NT, Dm, DFF, 1, 0, 0, 0, 0);
            decompose(xb2, xbuf, t3, Dm);
            k_add3_h<<<gridFor(ND), 256, 0, stream>>>(t1, t2, t3, q, ND);   // q reused as tsum
            k_im2col3<<<gridFor(NT * (long long)(3 * Dm)), 256, 0, stream>>>(q, big, Bc, L, Dm);
            gemm(big, dTwT[li], nullptr, trA, trB, (int)NT, CIN, 3 * Dm, 1, 0, 0, 0, 0);
            h16* tmp = trA; trA = trB; trB = tmp;
        }
        myln(xbuf, dec[li].ng, dec[li].nb, lnT);
        gemm(lnT, dPrT[li], dec[li].pb, nullptr, sOut, (int)NT, CIN, Dm, 1, 0, 0, 0, 0);
        k_accum_out<<<gridFor((long long)Bc * 512 * CIN), 256, 0, stream>>>(
            sOut, trA, (float*)d_out, Bc, 512, CIN, L, 0.5f);
    }
}
